// Decoder_18760417149687
// MI455X (gfx1250) — matrix operand order repaired and checked by reading, not compiled, not run on hardware
//
#include <hip/hip_runtime.h>
#include <stdint.h>

// ---------------------------------------------------------------------------
// Types for CDNA5 WMMA (gfx1250, wave32)
// ---------------------------------------------------------------------------
typedef __attribute__((ext_vector_type(16))) __bf16 v16bf;
typedef __attribute__((ext_vector_type(8)))  float  v8f;

union ABFrag { v16bf v; uint4 q[2]; };

// Problem constants
#define NB   64        // batch
#define TENC 512       // encoder length
#define TDEC 99        // decode steps (T_dec - 1)
#define HD   512
#define KS   128
#define VS   128
#define VOC  32000
#define K1   1152      // 640 + 512  (LSTM1 fused K)
#define K2   640       // 512 + 128  (LSTM2 fused K)
#define KO   256       // 128 + 128  (output proj K)
#define PRED_STRIDE ((size_t)TDEC * VOC)   // per-batch-row stride in predictions
#define ATT_STRIDE  ((size_t)TDEC * TENC)  // per-batch-row stride in attention maps

// ---------------------------------------------------------------------------
// bf16 WMMA GEMM:  C[m, j] = sum_k A[m,k] * W[j,k] + bias[j]
// A: [64 x K] bf16 row-major (batch),  W: [Ncols x K] bf16 row-major.
// blockDim = 128 (4 waves). blockIdx.y = batch tile (16 rows, staged in LDS,
// shared by all waves). Each wave computes FOUR adjacent 16x16 D tiles
// (16 x 64 output), reusing one A fragment across 4 WMMAs per K-chunk:
//   per K-chunk: 2 ds_load_b128 (A) + 8 global_load_b128 (B) + 4 v_wmma.
// Ncols must be a multiple of 256. A/B use contiguous k (16*half + s) per lane; D layout per CDNA5 ISA 7.12.2.
// ---------------------------------------------------------------------------
__global__ __launch_bounds__(128) void wmma_gemm_bf16(
    const __bf16* __restrict__ A,
    const __bf16* __restrict__ W,
    const float*  __restrict__ bias,
    float* __restrict__ C,
    long long cRowStride,      // elements between consecutive batch rows of C
    int K)                     // multiple of 32
{
    extern __shared__ __align__(16) unsigned char smem_raw[];
    __bf16* shA = (__bf16*)smem_raw;   // 16 rows x K bf16

    const int lane = threadIdx.x & 31;
    const int wave = threadIdx.x >> 5;
    const int half = lane >> 4;        // 0: lanes 0-15, 1: lanes 16-31
    const int lr   = lane & 15;
    const int m0   = blockIdx.y * 16;
    const int j0   = (blockIdx.x * 4 + wave) * 64;   // 4 N-tiles per wave

    // Cooperative stage of the shared A batch tile (16 x K) into LDS.
    const int kChunks = K >> 3;                  // 16-byte chunks per row
    for (int i = threadIdx.x; i < 16 * kChunks; i += 128) {
        const int row = i / kChunks;
        const int cc  = i - row * kChunks;
        *(uint4*)(shA + (size_t)row * K + cc * 8) =
            *(const uint4*)(A + (size_t)(m0 + row) * K + cc * 8);
    }
    __syncthreads();

    const __bf16* arow  = shA + (size_t)lr * K;             // batch row for this lane
    const __bf16* wrow0 = W + (size_t)(j0 +  0 + lr) * K;   // weight rows (out cols)
    const __bf16* wrow1 = W + (size_t)(j0 + 16 + lr) * K;
    const __bf16* wrow2 = W + (size_t)(j0 + 32 + lr) * K;
    const __bf16* wrow3 = W + (size_t)(j0 + 48 + lr) * K;

    v8f acc0 = {}, acc1 = {}, acc2 = {}, acc3 = {};
    for (int k0 = 0; k0 < K; k0 += 32) {
        ABFrag a, b0, b1, b2, b3;
        // A 16x32 bf16: lanes 0-15 -> K {0..15}; lanes 16-31 -> K {16..31}
        a.q[0] = *(const uint4*)(arow + k0 + half * 16);
        a.q[1] = *(const uint4*)(arow + k0 + half * 16 + 8);
        // B 32x16 bf16: lanes 0-15 -> K {0..15}; lanes 16-31 -> K {16..31}
        const int kb = k0 + half * 16;
        b0.q[0] = *(const uint4*)(wrow0 + kb);
        b0.q[1] = *(const uint4*)(wrow0 + kb + 8);
        b1.q[0] = *(const uint4*)(wrow1 + kb);
        b1.q[1] = *(const uint4*)(wrow1 + kb + 8);
        b2.q[0] = *(const uint4*)(wrow2 + kb);
        b2.q[1] = *(const uint4*)(wrow2 + kb + 8);
        b3.q[0] = *(const uint4*)(wrow3 + kb);
        b3.q[1] = *(const uint4*)(wrow3 + kb + 8);
        acc0 = __builtin_amdgcn_wmma_f32_16x16x32_bf16(false, a.v, false, b0.v,
                                                       (short)0, acc0, false, false);
        acc1 = __builtin_amdgcn_wmma_f32_16x16x32_bf16(false, a.v, false, b1.v,
                                                       (short)0, acc1, false, false);
        acc2 = __builtin_amdgcn_wmma_f32_16x16x32_bf16(false, a.v, false, b2.v,
                                                       (short)0, acc2, false, false);
        acc3 = __builtin_amdgcn_wmma_f32_16x16x32_bf16(false, a.v, false, b3.v,
                                                       (short)0, acc3, false, false);
    }

    const float bj0 = bias ? bias[j0 +  0 + lr] : 0.0f;
    const float bj1 = bias ? bias[j0 + 16 + lr] : 0.0f;
    const float bj2 = bias ? bias[j0 + 32 + lr] : 0.0f;
    const float bj3 = bias ? bias[j0 + 48 + lr] : 0.0f;
    // D layout: VGPR r -> row m0+r (lanes 0-15) / m0+r+8 (lanes 16-31), col tile+(lane&15)
    #pragma unroll
    for (int r = 0; r < 8; ++r) {
        const size_t rowOff = (size_t)(m0 + r + half * 8) * (size_t)cRowStride;
        C[rowOff + j0 +  0 + lr] = acc0[r] + bj0;
        C[rowOff + j0 + 16 + lr] = acc1[r] + bj1;
        C[rowOff + j0 + 32 + lr] = acc2[r] + bj2;
        C[rowOff + j0 + 48 + lr] = acc3[r] + bj3;
    }
}

// ---------------------------------------------------------------------------
// Elementwise / setup kernels
// ---------------------------------------------------------------------------
__device__ __forceinline__ float sigm(float x) { return 1.0f / (1.0f + expf(-x)); }

// LSTM cell 1 activation: gates [64,2048] (i,f,g,o x 512), state c1 [64,512]
__global__ void lstm_act1(const float* __restrict__ g, float* __restrict__ c,
                          __bf16* __restrict__ a2_h1,       // A2 h1 slot (stride 640)
                          __bf16* __restrict__ a1h1_next)   // A1[t+1]+640 (stride 1152) or null
{
    const int idx = blockIdx.x * blockDim.x + threadIdx.x;  // 64*512 threads
    const int n = idx >> 9, u = idx & 511;
    const float* gr = g + (size_t)n * 2048;
    const float gi = gr[u], gf = gr[512 + u], gg = gr[1024 + u], go = gr[1536 + u];
    const float cn = sigm(gf) * c[idx] + sigm(gi) * tanhf(gg);
    const float h  = sigm(go) * tanhf(cn);
    c[idx] = cn;
    const __bf16 hb = (__bf16)h;
    a2_h1[(size_t)n * K2 + u] = hb;
    if (a1h1_next) a1h1_next[(size_t)n * K1 + u] = hb;
}

// LSTM cell 2 activation: gates [64,512] (i,f,g,o x 128), state c2 [64,128]
__global__ void lstm_act2(const float* __restrict__ g, float* __restrict__ c,
                          float*  __restrict__ h2f,     // [64,128]
                          __bf16* __restrict__ a2_h2,   // A2+512 (stride 640)
                          __bf16* __restrict__ x_h2)    // X base (stride 256)
{
    const int idx = blockIdx.x * blockDim.x + threadIdx.x;  // 64*128 threads
    const int n = idx >> 7, u = idx & 127;
    const float* gr = g + (size_t)n * 512;
    const float gi = gr[u], gf = gr[128 + u], gg = gr[256 + u], go = gr[384 + u];
    const float cn = sigm(gf) * c[idx] + sigm(gi) * tanhf(gg);
    const float h  = sigm(go) * tanhf(cn);
    c[idx] = cn;
    h2f[idx] = h;
    const __bf16 hb = (__bf16)h;
    a2_h2[(size_t)n * K2 + u] = hb;
    x_h2[(size_t)n * KO + u]  = hb;
}

// Per-batch attention: energies, masked softmax (== softmax*mask/renorm),
// attention-map output, and context = attn @ values.
__global__ __launch_bounds__(256) void attention_kernel(
    const float* __restrict__ enc_key,   // [64,512,128]
    const float* __restrict__ values,    // [64,512,128]
    const int*   __restrict__ lens,      // [64]
    const float* __restrict__ h2f,       // [64,128]
    float* __restrict__ attn_out,        // + t*512; row stride ATT_STRIDE
    __bf16* __restrict__ x_ctx,          // X + 128 (stride 256)
    __bf16* __restrict__ a1ctx_next)     // A1[t+1]+512 (stride 1152) or null
{
    __shared__ float sh_h2[KS];
    __shared__ float sh_e[TENC];
    __shared__ float rbuf[256];

    const int n = blockIdx.x;
    const int tid = threadIdx.x;
    if (tid < KS) sh_h2[tid] = h2f[(size_t)n * KS + tid];
    __syncthreads();

    // energies: e[t] = enc_key[n,t,:] . h2
    for (int t = tid; t < TENC; t += 256) {
        const float4* kp = (const float4*)(enc_key + ((size_t)n * TENC + t) * KS);
        float e = 0.f;
        #pragma unroll 4
        for (int k4 = 0; k4 < KS / 4; ++k4) {
            const float4 kv = kp[k4];
            e += kv.x * sh_h2[4 * k4] + kv.y * sh_h2[4 * k4 + 1] +
                 kv.z * sh_h2[4 * k4 + 2] + kv.w * sh_h2[4 * k4 + 3];
        }
        sh_e[t] = e;
    }
    __syncthreads();

    // max over ALL t (matches reference softmax numerics)
    float m = fmaxf(sh_e[tid], sh_e[tid + 256]);
    rbuf[tid] = m; __syncthreads();
    for (int s = 128; s > 0; s >>= 1) {
        if (tid < s) rbuf[tid] = fmaxf(rbuf[tid], rbuf[tid + s]);
        __syncthreads();
    }
    const float gmax = rbuf[0];
    __syncthreads();

    // exp + mask + sum
    const int L = lens[n];
    float lsum = 0.f;
    for (int t = tid; t < TENC; t += 256) {
        float ev = expf(sh_e[t] - gmax);
        if (t >= L) ev = 0.f;
        sh_e[t] = ev;
        lsum += ev;
    }
    __syncthreads();
    rbuf[tid] = lsum; __syncthreads();
    for (int s = 128; s > 0; s >>= 1) {
        if (tid < s) rbuf[tid] += rbuf[tid + s];
        __syncthreads();
    }
    const float inv = 1.0f / rbuf[0];

    for (int t = tid; t < TENC; t += 256) {
        const float a = sh_e[t] * inv;
        sh_e[t] = a;
        attn_out[(size_t)n * ATT_STRIDE + t] = a;
    }
    __syncthreads();

    // context[v] = sum_t attn[t] * values[n,t,v]
    if (tid < VS) {
        float acc = 0.f;
        for (int t = 0; t < TENC; ++t)
            acc += sh_e[t] * values[((size_t)n * TENC + t) * VS + tid];
        const __bf16 cb = (__bf16)acc;
        x_ctx[(size_t)n * KO + tid] = cb;
        if (a1ctx_next) a1ctx_next[(size_t)n * K1 + tid] = cb;
    }
}

// Embedding gather for all steps: A1[t][n][0:512] = bf16(embedding[text[n,t]])
__global__ void embed_all(const int* __restrict__ text, const float* __restrict__ emb,
                          __bf16* __restrict__ A1)
{
    const int idx = blockIdx.x * blockDim.x + threadIdx.x;  // TDEC*64*512
    const int u = idx & 511;
    const int n = (idx >> 9) & 63;
    const int t = idx >> 15;
    if (t >= TDEC) return;
    const int tok = text[n * 100 + t];     // text is [64,100]; steps use t=0..98
    A1[((size_t)t * NB + n) * K1 + u] = (__bf16)emb[(size_t)tok * HD + u];
}

// Fused-weight conversions to bf16
__global__ void cvt_w1(const float* __restrict__ Wih1, const float* __restrict__ Whh1,
                       __bf16* __restrict__ Wc)
{
    const size_t idx = (size_t)blockIdx.x * blockDim.x + threadIdx.x;  // 2048*1152
    if (idx >= (size_t)2048 * K1) return;
    const int j = (int)(idx / K1), k = (int)(idx % K1);
    const float v = (k < 640) ? Wih1[(size_t)j * 640 + k] : Whh1[(size_t)j * 512 + (k - 640)];
    Wc[idx] = (__bf16)v;
}
__global__ void cvt_w2(const float* __restrict__ Wih2, const float* __restrict__ Whh2,
                       __bf16* __restrict__ Wc)
{
    const size_t idx = (size_t)blockIdx.x * blockDim.x + threadIdx.x;  // 512*640
    if (idx >= (size_t)512 * K2) return;
    const int j = (int)(idx / K2), k = (int)(idx % K2);
    const float v = (k < 512) ? Wih2[(size_t)j * 512 + k] : Whh2[(size_t)j * 128 + (k - 512)];
    Wc[idx] = (__bf16)v;
}
__global__ void cvt_f32_bf16(const float* __restrict__ src, __bf16* __restrict__ dst, size_t n)
{
    const size_t i = (size_t)blockIdx.x * blockDim.x + threadIdx.x;
    if (i < n) dst[i] = (__bf16)src[i];
}
__global__ void add_bias(const float* __restrict__ a, const float* __restrict__ b,
                         float* __restrict__ o, int n)
{
    const int i = blockIdx.x * blockDim.x + threadIdx.x;
    if (i < n) o[i] = a[i] + b[i];
}
__global__ void zero_f32(float* p, int n)
{
    const int i = blockIdx.x * blockDim.x + threadIdx.x;
    if (i < n) p[i] = 0.f;
}
__global__ void zero_bf16_slot(__bf16* p, int rows, int rowStride, int off, int len)
{
    const int i = blockIdx.x * blockDim.x + threadIdx.x;
    if (i < rows * len) {
        const int r = i / len, c = i - r * len;
        p[(size_t)r * rowStride + off + c] = (__bf16)0.f;
    }
}

// ---------------------------------------------------------------------------
// Host launcher
// ---------------------------------------------------------------------------
extern "C" void kernel_launch(void* const* d_in, const int* in_sizes, int n_in,
                              void* d_out, int out_size, void* d_ws, size_t ws_size,
                              hipStream_t stream)
{
    const float* enc_key   = (const float*)d_in[0];
    const float* values    = (const float*)d_in[1];
    const int*   lens      = (const int*)d_in[2];
    const int*   text      = (const int*)d_in[3];
    const float* embedding = (const float*)d_in[4];
    const float* Wih1 = (const float*)d_in[5];
    const float* Whh1 = (const float*)d_in[6];
    const float* bih1 = (const float*)d_in[7];
    const float* bhh1 = (const float*)d_in[8];
    const float* Wih2 = (const float*)d_in[9];
    const float* Whh2 = (const float*)d_in[10];
    const float* bih2 = (const float*)d_in[11];
    const float* bhh2 = (const float*)d_in[12];
    const float* Wout = (const float*)d_in[13];
    const float* bout = (const float*)d_in[14];

    float* pred_base = (float*)d_out;                                   // [64][99][32000]
    float* attn_base = (float*)d_out + (size_t)NB * TDEC * VOC;         // [64][99][512]

    // Carve workspace (256B aligned slabs)
    uintptr_t p = (uintptr_t)d_ws;
    auto carve = [&](size_t bytes) -> void* {
        uintptr_t q = (p + 255) & ~(uintptr_t)255;
        p = q + bytes;
        return (void*)q;
    };
    __bf16* W1c  = (__bf16*)carve((size_t)2048 * K1 * 2);   // fused [Wih1|Whh1]
    __bf16* W2c  = (__bf16*)carve((size_t)512  * K2 * 2);   // fused [Wih2|Whh2]
    __bf16* Wob  = (__bf16*)carve((size_t)VOC  * KO * 2);   // Wout bf16
    float*  b1   = (float*)carve(2048 * 4);
    float*  b2   = (float*)carve(512 * 4);
    __bf16* A1   = (__bf16*)carve((size_t)TDEC * NB * K1 * 2); // [t][n][x|ctx|h1]
    __bf16* A2   = (__bf16*)carve((size_t)NB * K2 * 2);        // [n][h1|h2]
    __bf16* X    = (__bf16*)carve((size_t)NB * KO * 2);        // [n][h2|ctx]
    float*  g1   = (float*)carve((size_t)NB * 2048 * 4);
    float*  g2   = (float*)carve((size_t)NB * 512 * 4);
    float*  c1   = (float*)carve((size_t)NB * HD * 4);
    float*  c2   = (float*)carve((size_t)NB * KS * 4);
    float*  h2f  = (float*)carve((size_t)NB * KS * 4);

    // ---- one-time setup (per launch; deterministic, capture-safe) ----
    cvt_w1<<<(2048 * K1 + 255) / 256, 256, 0, stream>>>(Wih1, Whh1, W1c);
    cvt_w2<<<(512 * K2 + 255) / 256, 256, 0, stream>>>(Wih2, Whh2, W2c);
    cvt_f32_bf16<<<((size_t)VOC * KO + 255) / 256, 256, 0, stream>>>(Wout, Wob, (size_t)VOC * KO);
    add_bias<<<8, 256, 0, stream>>>(bih1, bhh1, b1, 2048);
    add_bias<<<2, 256, 0, stream>>>(bih2, bhh2, b2, 512);
    embed_all<<<(TDEC * NB * HD + 255) / 256, 256, 0, stream>>>(text, embedding, A1);
    zero_f32<<<(NB * HD + 255) / 256, 256, 0, stream>>>(c1, NB * HD);
    zero_f32<<<(NB * KS + 255) / 256, 256, 0, stream>>>(c2, NB * KS);
    // A1[0] ctx+h1 slots (cols 512..1151) and A2 h2 slot start at zero
    zero_bf16_slot<<<(NB * 640 + 255) / 256, 256, 0, stream>>>(A1, NB, K1, 512, 640);
    zero_bf16_slot<<<(NB * 128 + 255) / 256, 256, 0, stream>>>(A2, NB, K2, 512, 128);

    // ---- 99 sequential decode steps ----
    for (int t = 0; t < TDEC; ++t) {
        __bf16* a1_t    = A1 + (size_t)t * NB * K1;
        __bf16* a1_next = (t + 1 < TDEC) ? A1 + (size_t)(t + 1) * NB * K1 : nullptr;

        // LSTM1 fused GEMM: gates1 = [x|ctx|h1] @ [Wih1|Whh1]^T + b1
        wmma_gemm_bf16<<<dim3(2048 / 256, 4), 128, 16 * K1 * 2, stream>>>(
            a1_t, W1c, b1, g1, 2048, K1);
        lstm_act1<<<(NB * HD) / 256, 256, 0, stream>>>(
            g1, c1, A2, a1_next ? a1_next + 640 : nullptr);

        // LSTM2 fused GEMM: gates2 = [h1|h2] @ [Wih2|Whh2]^T + b2
        wmma_gemm_bf16<<<dim3(512 / 256, 4), 128, 16 * K2 * 2, stream>>>(
            A2, W2c, b2, g2, 512, K2);
        lstm_act2<<<(NB * KS) / 256, 256, 0, stream>>>(g2, c2, h2f, A2 + 512, X);

        // Attention + context (writes attention map for this step)
        attention_kernel<<<NB, 256, 0, stream>>>(
            enc_key, values, lens, h2f,
            attn_base + (size_t)t * TENC, X + 128,
            a1_next ? a1_next + 512 : nullptr);

        // Output projection: preds[:, t, :] = [h2|ctx] @ Wout^T + bout
        wmma_gemm_bf16<<<dim3(VOC / 256, 4), 128, 16 * KO * 2, stream>>>(
            X, Wob, bout, pred_base + (size_t)t * VOC, (long long)PRED_STRIDE, KO);
    }
}
